// SwinTransformerBlock_824633721562
// MI455X (gfx1250) — matrix operand order repaired and checked by reading, not compiled, not run on hardware
//
#include <hip/hip_runtime.h>
#include <hip/hip_bf16.h>
#include <stdint.h>

typedef _Float16 half_t;
typedef __attribute__((ext_vector_type(16))) _Float16 v16h;
typedef __attribute__((ext_vector_type(8)))  _Float16 v8h;
typedef __attribute__((ext_vector_type(8)))  float    v8f;

// ---------------- problem constants ----------------
constexpr int Bsz   = 16;
constexpr int Himg  = 56;
constexpr int Wimg  = 56;
constexpr int Cc    = 512;
constexpr int HEADS = 16;
constexpr int HD    = 32;          // head dim
constexpr int SHF   = 3;
constexpr int NTOK  = 49;          // tokens per window
constexpr int NWIN  = 64;          // windows per image (8x8)
constexpr int WINS  = Bsz * NWIN;  // 1024 windows total
constexpr int TOKENS = WINS * NTOK; // 50176 rows
constexpr int HID   = 2048;
constexpr float QSCALE = 0.17677669529663687f; // 32^-0.5

// ---------------- WMMA helper ----------------
__device__ inline v8f wmma_f16(v16h a, v16h b, v8f c) {
  // v_wmma_f32_16x16x32_f16  D = A(16x32) * B(32x16) + C
  return __builtin_amdgcn_wmma_f32_16x16x32_f16(false, a, false, b, (short)0, c,
                                                false, false);
}

__device__ inline v16h make_v16(v8h lo, v8h hi) {
  union { v16h v; v8h h[2]; } u;
  u.h[0] = lo; u.h[1] = hi;
  return u.v;
}

// ---------------- async global -> LDS copy (CDNA5, ASYNCcnt-tracked) ----------
__device__ inline void async_copy_b128(uint32_t lds_off, const half_t* g) {
  // LDS address comes from the vdst VGPR; generic shared-pointer low 32 bits = LDS offset
  asm volatile("global_load_async_to_lds_b128 %0, %1, off"
               :: "v"(lds_off), "v"((uint64_t)(uintptr_t)g)
               : "memory");
}

// ---------------- fp32 -> fp16 convert ----------------
__global__ void f32_to_f16_kernel(const float* __restrict__ in,
                                  half_t* __restrict__ out, int n) {
  int i = blockIdx.x * blockDim.x + threadIdx.x;
  if (i < n) out[i] = (half_t)in[i];
}

// ---------------- LayerNorm (optionally fused shift + window partition) ----------
__global__ __launch_bounds__(256)
void layernorm_kernel(const float* __restrict__ x, const float* __restrict__ g,
                      const float* __restrict__ bta, half_t* __restrict__ out,
                      int do_shift) {
  const int t = blockIdx.x;
  int src;
  if (do_shift) {
    int win = t / NTOK, n = t - win * NTOK;
    int b  = win >> 6, wi = win & 63;
    int wh = wi >> 3,  ww = wi & 7;
    int rr = n / 7,    cw = n - rr * 7;
    int hh = wh * 7 + rr + SHF; if (hh >= Himg) hh -= Himg;
    int w0 = ww * 7 + cw + SHF; if (w0 >= Wimg) w0 -= Wimg;
    src = b * (Himg * Wimg) + hh * Wimg + w0;
  } else {
    src = t;
  }
  const float* xr = x + (size_t)src * Cc;
  const int c0 = threadIdx.x * 2;
  float v0 = xr[c0], v1 = xr[c0 + 1];
  float s = v0 + v1, q = v0 * v0 + v1 * v1;
#pragma unroll
  for (int o = 16; o > 0; o >>= 1) {
    s += __shfl_xor(s, o);
    q += __shfl_xor(q, o);
  }
  __shared__ float ss[8], sq[8];
  const int wv = threadIdx.x >> 5;
  if ((threadIdx.x & 31) == 0) { ss[wv] = s; sq[wv] = q; }
  __syncthreads();
  if (threadIdx.x == 0) {
    float S = 0.f, Q = 0.f;
#pragma unroll
    for (int i = 0; i < 8; i++) { S += ss[i]; Q += sq[i]; }
    ss[0] = S; sq[0] = Q;
  }
  __syncthreads();
  const float mean = ss[0] * (1.0f / Cc);
  const float var  = sq[0] * (1.0f / Cc) - mean * mean;
  const float rs   = rsqrtf(var + 1e-5f);
  half_t* orow = out + (size_t)t * Cc;
  orow[c0]     = (half_t)((v0 - mean) * rs * g[c0]     + bta[c0]);
  orow[c0 + 1] = (half_t)((v1 - mean) * rs * g[c0 + 1] + bta[c0 + 1]);
}

// ---------------- WMMA GEMM with async-LDS double buffering ----------------
// out(M,N) = A(M,K) @ Wt(N,K)^T + bias, fused epilogue.
enum { EPI_QKV = 0, EPI_PROJ = 1, EPI_GELU = 2, EPI_FC2 = 3 };

template <int EPI>
__global__ __launch_bounds__(256)
void gemm_wmma_kernel(const half_t* __restrict__ A, const half_t* __restrict__ Wt,
                      const float* __restrict__ bias, int M, int Ncol, int K,
                      half_t* __restrict__ oh, float* __restrict__ of,
                      const float* __restrict__ res,
                      half_t* __restrict__ qb, half_t* __restrict__ kb,
                      half_t* __restrict__ vb) {
  // double-buffered tiles: A 128x32 f16 (8KB), B 64x32 f16 (4KB)
  __shared__ __align__(64) half_t shA[2][128 * 32];
  __shared__ __align__(64) half_t shB[2][64 * 32];

  const int tid  = threadIdx.x;
  const int lane = tid & 31;
  const int wave = tid >> 5;              // 8 waves: 4 along M, 2 along N
  const int wm = wave & 3, wn = wave >> 2;
  const int rowBase = blockIdx.y * 128;
  const int colBase = blockIdx.x * 64;
  const int row0 = rowBase + wm * 32;
  const int col0 = colBase + wn * 32;
  const int l15 = lane & 15;
  const int hb8  = (lane >> 4) * 8;
  const int hb16 = (lane >> 4) * 16;

  // per-thread copy assignments: A chunks {tid, tid+256}, B chunk {tid} (16B each)
  const int ra0 = tid >> 2,          oa0 = (tid & 3) * 8;
  const int ra1 = (tid >> 2) + 64,   oa1 = oa0;
  const int rb  = tid >> 2,          ob  = oa0;
  const half_t* gA0 = A  + (size_t)(rowBase + ra0) * K + oa0;
  const half_t* gA1 = A  + (size_t)(rowBase + ra1) * K + oa1;
  const half_t* gB  = Wt + (size_t)(colBase + rb ) * K + ob;
  uint32_t ldsA0[2], ldsA1[2], ldsB[2];
#pragma unroll
  for (int bse = 0; bse < 2; bse++) {
    ldsA0[bse] = (uint32_t)(uintptr_t)&shA[bse][(size_t)tid * 8];
    ldsA1[bse] = (uint32_t)(uintptr_t)&shA[bse][(size_t)(tid + 256) * 8];
    ldsB [bse] = (uint32_t)(uintptr_t)&shB[bse][(size_t)tid * 8];
  }

  auto issue = [&](int buf, int k0) {
    async_copy_b128(ldsA0[buf], gA0 + k0);
    async_copy_b128(ldsA1[buf], gA1 + k0);
    async_copy_b128(ldsB [buf], gB  + k0);
  };

  v8f acc00 = {}, acc01 = {}, acc10 = {}, acc11 = {};
  const int KT = K >> 5;
  issue(0, 0);
  for (int kt = 0; kt < KT; ++kt) {
    const int buf = kt & 1;
    if (kt + 1 < KT) {
      issue(buf ^ 1, (kt + 1) << 5);
      // 3 newest (next tile) may remain in flight; current tile complete
      asm volatile("s_wait_asynccnt 0x3" ::: "memory");
    } else {
      asm volatile("s_wait_asynccnt 0x0" ::: "memory");
    }
    __syncthreads();   // all waves' copies for this tile have landed

    const half_t* As0 = &shA[buf][(wm * 32 + l15) * 32 + hb8];
    const half_t* As1 = As0 + 16 * 32;
    v16h a0 = make_v16(*(const v8h*)As0, *(const v8h*)(As0 + 16));
    v16h a1 = make_v16(*(const v8h*)As1, *(const v8h*)(As1 + 16));
    const half_t* Bs0 = &shB[buf][(wn * 32 + l15) * 32 + hb8];
    const half_t* Bs1 = Bs0 + 16 * 32;
    v16h b0 = make_v16(*(const v8h*)Bs0, *(const v8h*)(Bs0 + 16));
    v16h b1 = make_v16(*(const v8h*)Bs1, *(const v8h*)(Bs1 + 16));
    acc00 = wmma_f16(a0, b0, acc00);
    acc01 = wmma_f16(a0, b1, acc01);
    acc10 = wmma_f16(a1, b0, acc10);
    acc11 = wmma_f16(a1, b1, acc11);
    __syncthreads();   // done reading buf before it is overwritten next round
  }

  auto epi = [&](int m, int n, float val) {
    val += bias[n];
    if (EPI == EPI_QKV) {
      int win = m / NTOK, tn = m - win * NTOK;
      int sec = n >> 9, cc = n & 511;          // 0:q 1:k 2:v
      int head = cc >> 5, d = cc & 31;
      size_t idx = (((size_t)win * HEADS + head) * 64 + tn) * HD + d;
      if (sec == 0)      qb[idx] = (half_t)(val * QSCALE);
      else if (sec == 1) kb[idx] = (half_t)val;
      else vb[(((size_t)win * HEADS + head) * HD + d) * 64 + tn] = (half_t)val; // V^T
    } else if (EPI == EPI_PROJ) {
      int win = m / NTOK, tn = m - win * NTOK;
      int b  = win >> 6, wi = win & 63;
      int wh = wi >> 3,  ww = wi & 7;
      int rr = tn / 7,   cw = tn - rr * 7;
      int hh = wh * 7 + rr + SHF; if (hh >= Himg) hh -= Himg;
      int w0 = ww * 7 + cw + SHF; if (w0 >= Wimg) w0 -= Wimg;
      size_t dst = ((size_t)b * (Himg * Wimg) + hh * Wimg + w0) * Cc + n;
      of[dst] = res[dst] + val;
    } else if (EPI == EPI_GELU) {
      float gch = 0.5f * val * (1.0f + erff(val * 0.70710678118654752f));
      oh[(size_t)m * Ncol + n] = (half_t)gch;
    } else { // EPI_FC2: residual add, fp32 out
      size_t dst = (size_t)m * Cc + n;
      of[dst] = res[dst] + val;
    }
  };

#pragma unroll
  for (int e = 0; e < 8; e++) {
    int mb = e + hb8;
    epi(row0 + mb,      col0 + l15,      acc00[e]);
    epi(row0 + mb,      col0 + 16 + l15, acc01[e]);
    epi(row0 + 16 + mb, col0 + l15,      acc10[e]);
    epi(row0 + 16 + mb, col0 + 16 + l15, acc11[e]);
  }
}

// ---------------- attention: 4 waves per (window, head) ----------------
__global__ __launch_bounds__(128)
void attn_kernel(const half_t* __restrict__ q, const half_t* __restrict__ k,
                 const half_t* __restrict__ vt, const float* __restrict__ rpb,
                 half_t* __restrict__ o16) {
  __shared__ float  sc[64][65];                 // padded rows: conflict-free
  __shared__ __align__(16) half_t at[64][72];   // 144B rows: 16B-aligned frags
  __shared__ float  bl[169];                    // rel-pos bias for this head

  const int head = blockIdx.x, win = blockIdx.y;
  const int tid  = threadIdx.x;
  const int lane = tid & 31;
  const int mt   = tid >> 5;                    // wave id = row tile
  const int l15  = lane & 15;
  const int hb8  = (lane >> 4) * 8;
  const int hb16 = (lane >> 4) * 16;

  for (int i = tid; i < 169; i += 128) bl[i] = rpb[i * HEADS + head];

  const half_t* qp = q  + ((size_t)win * HEADS + head) * 64 * HD;
  const half_t* kp = k  + ((size_t)win * HEADS + head) * 64 * HD;
  const half_t* vp = vt + ((size_t)win * HEADS + head) * HD * 64;

  // ---- scores = (scaled Q) @ K^T : wave mt computes 16x64 strip ----
  {
    const half_t* ap = qp + (size_t)(mt * 16 + l15) * HD + hb8;
    v16h af = make_v16(*(const v8h*)ap, *(const v8h*)(ap + 16));
#pragma unroll
    for (int nt = 0; nt < 4; nt++) {
      const half_t* bp = kp + (size_t)(nt * 16 + l15) * HD + hb8;
      v16h bf = make_v16(*(const v8h*)bp, *(const v8h*)(bp + 16));
      v8f accv = {};
      accv = wmma_f16(af, bf, accv);
#pragma unroll
      for (int e = 0; e < 8; e++)
        sc[mt * 16 + e + hb8][nt * 16 + l15] = accv[e];
    }
  }
  __syncthreads();

  // ---- softmax (bias + shifted-window mask): lanes 0-15, one row each ----
  const int wi = win & 63, wh = wi >> 3, ww = wi & 7;
  if (lane < 16) {
    int r = mt * 16 + lane;
    if (r < NTOK) {
      int r1 = r / 7, c1 = r - r1 * 7;
      int hsr = wh * 7 + r1, wsr = ww * 7 + c1;   // shifted-map coords
      int regr = ((hsr < 49) ? 0 : ((hsr < 53) ? 1 : 2)) * 3 +
                 ((wsr < 49) ? 0 : ((wsr < 53) ? 1 : 2));
      float mx = -1e30f;
      int r2 = 0, c2 = 0;
      for (int m = 0; m < NTOK; m++) {
        int hsm = wh * 7 + r2, wsm = ww * 7 + c2;
        int regm = ((hsm < 49) ? 0 : ((hsm < 53) ? 1 : 2)) * 3 +
                   ((wsm < 49) ? 0 : ((wsm < 53) ? 1 : 2));
        float sv = sc[r][m] + bl[(r1 - r2 + 6) * 13 + (c1 - c2 + 6)] +
                   ((regm != regr) ? -100.0f : 0.0f);
        sc[r][m] = sv;
        mx = fmaxf(mx, sv);
        if (++c2 == 7) { c2 = 0; ++r2; }
      }
      float sum = 0.f;
      for (int m = 0; m < NTOK; m++) {
        float ev = __expf(sc[r][m] - mx);
        sc[r][m] = ev;
        sum += ev;
      }
      float inv = 1.0f / sum;
      for (int m = 0; m < NTOK; m++) at[r][m] = (half_t)(sc[r][m] * inv);
      for (int m = NTOK; m < 64; m++) at[r][m] = (half_t)0.f; // neutralize pad
    } else {
      for (int m = 0; m < 64; m++) at[r][m] = (half_t)0.f;
    }
  }
  __syncthreads();

  // ---- out = attn(64x64) @ V(64x32) via V^T rows as B columns ----
  {
    const half_t* a0p = &at[mt * 16 + l15][hb8];
    const half_t* a1p = &at[mt * 16 + l15][32 + hb8];
    v16h a0 = make_v16(*(const v8h*)a0p, *(const v8h*)(a0p + 16));
    v16h a1 = make_v16(*(const v8h*)a1p, *(const v8h*)(a1p + 16));
#pragma unroll
    for (int nt = 0; nt < 2; nt++) {
      const half_t* b0p = vp + (size_t)(nt * 16 + l15) * 64 + hb8;
      v16h b0 = make_v16(*(const v8h*)b0p, *(const v8h*)(b0p + 16));
      v16h b1 = make_v16(*(const v8h*)(b0p + 32), *(const v8h*)(b0p + 48));
      v8f accv = {};
      accv = wmma_f16(a0, b0, accv);
      accv = wmma_f16(a1, b1, accv);
#pragma unroll
      for (int e = 0; e < 8; e++) {
        int row = mt * 16 + e + hb8;
        if (row < NTOK)
          o16[((size_t)win * NTOK + row) * Cc + head * HD + nt * 16 + l15] =
              (half_t)accv[e];
      }
    }
  }
}

// ---------------- launch ----------------
extern "C" void kernel_launch(void* const* d_in, const int* in_sizes, int n_in,
                              void* d_out, int out_size, void* d_ws, size_t ws_size,
                              hipStream_t stream) {
  const float* x      = (const float*)d_in[0];
  const float* n1g    = (const float*)d_in[1];
  const float* n1b    = (const float*)d_in[2];
  const float* qkv_w  = (const float*)d_in[3];
  const float* qkv_b  = (const float*)d_in[4];
  const float* proj_w = (const float*)d_in[5];
  const float* proj_b = (const float*)d_in[6];
  const float* rpb    = (const float*)d_in[7];
  const float* n2g    = (const float*)d_in[8];
  const float* n2b    = (const float*)d_in[9];
  const float* fc1_w  = (const float*)d_in[10];
  const float* fc1_b  = (const float*)d_in[11];
  const float* fc2_w  = (const float*)d_in[12];
  const float* fc2_b  = (const float*)d_in[13];
  float* out = (float*)d_out;

  char* ws = (char*)d_ws;
  size_t off = 0;
  auto wsalloc = [&](size_t bytes) -> char* {
    char* p = ws + off;
    off += (bytes + 255) & ~(size_t)255;
    return p;
  };
  half_t* h16   = (half_t*)wsalloc((size_t)TOKENS * Cc * 2);
  half_t* qb    = (half_t*)wsalloc((size_t)WINS * HEADS * 64 * HD * 2);
  half_t* kb    = (half_t*)wsalloc((size_t)WINS * HEADS * 64 * HD * 2);
  half_t* vtb   = (half_t*)wsalloc((size_t)WINS * HEADS * HD * 64 * 2);
  half_t* o16   = (half_t*)wsalloc((size_t)TOKENS * Cc * 2);
  float*  x1    = (float*) wsalloc((size_t)TOKENS * Cc * 4);
  half_t* m16   = (half_t*)wsalloc((size_t)TOKENS * Cc * 2);
  half_t* g16   = (half_t*)wsalloc((size_t)TOKENS * HID * 2);
  half_t* wqkv  = (half_t*)wsalloc((size_t)3 * Cc * Cc * 2);
  half_t* wproj = (half_t*)wsalloc((size_t)Cc * Cc * 2);
  half_t* wfc1  = (half_t*)wsalloc((size_t)HID * Cc * 2);
  half_t* wfc2  = (half_t*)wsalloc((size_t)Cc * HID * 2);

  // fp16 weight copies (tiny, L2-resident)
  int nw;
  nw = 3 * Cc * Cc; f32_to_f16_kernel<<<(nw + 255) / 256, 256, 0, stream>>>(qkv_w, wqkv, nw);
  nw = Cc * Cc;     f32_to_f16_kernel<<<(nw + 255) / 256, 256, 0, stream>>>(proj_w, wproj, nw);
  nw = HID * Cc;    f32_to_f16_kernel<<<(nw + 255) / 256, 256, 0, stream>>>(fc1_w, wfc1, nw);
  nw = Cc * HID;    f32_to_f16_kernel<<<(nw + 255) / 256, 256, 0, stream>>>(fc2_w, wfc2, nw);

  // LN1 + cyclic shift + window partition -> f16
  layernorm_kernel<<<TOKENS, 256, 0, stream>>>(x, n1g, n1b, h16, 1);

  // QKV projection (scatter q/k/v^T, q pre-scaled)
  gemm_wmma_kernel<EPI_QKV><<<dim3((3 * Cc) / 64, TOKENS / 128), 256, 0, stream>>>(
      h16, wqkv, qkv_b, TOKENS, 3 * Cc, Cc, nullptr, nullptr, nullptr, qb, kb, vtb);

  // windowed attention with rel-pos bias + shift mask
  attn_kernel<<<dim3(HEADS, WINS), 128, 0, stream>>>(qb, kb, vtb, rpb, o16);

  // output projection + window reverse + un-shift + residual -> x1 (fp32)
  gemm_wmma_kernel<EPI_PROJ><<<dim3(Cc / 64, TOKENS / 128), 256, 0, stream>>>(
      o16, wproj, proj_b, TOKENS, Cc, Cc, nullptr, x1, x, nullptr, nullptr, nullptr);

  // LN2 -> f16
  layernorm_kernel<<<TOKENS, 256, 0, stream>>>(x1, n2g, n2b, m16, 0);

  // fc1 + exact GELU -> f16
  gemm_wmma_kernel<EPI_GELU><<<dim3(HID / 64, TOKENS / 128), 256, 0, stream>>>(
      m16, wfc1, fc1_b, TOKENS, HID, Cc, g16, nullptr, nullptr, nullptr, nullptr, nullptr);

  // fc2 + residual -> fp32 output
  gemm_wmma_kernel<EPI_FC2><<<dim3(Cc / 64, TOKENS / 128), 256, 0, stream>>>(
      g16, wfc2, fc2_b, TOKENS, Cc, HID, nullptr, out, x1, nullptr, nullptr, nullptr);
}
